// LE2EBlock_26903675142582
// MI455X (gfx1250) — matrix operand order repaired and checked by reading, not compiled, not run on hardware
//
#include <hip/hip_runtime.h>
#include <math.h>

// ---------------- problem constants ----------------
#define BB    8
#define TT    1024
#define DIMC  512
#define HEADS 4
#define HEADD 128          // DIM / HEADS
#define FFNC  2048         // 4 * DIM
#define MTOT  (BB * TT)    // 8192 rows

typedef __attribute__((ext_vector_type(16))) __bf16 bfx16;
typedef __attribute__((ext_vector_type(8)))  __bf16 bfx8;
typedef __attribute__((ext_vector_type(8)))  float  v8f;

// ---------------- device helpers ----------------
__device__ __forceinline__ v8f wmma_bf16(const bfx16 a, const bfx16 b, const v8f c) {
    return __builtin_amdgcn_wmma_f32_16x16x32_bf16(false, a, false, b, (short)0, c, false, false);
}

__device__ __forceinline__ bfx16 mkfrag(const bfx8 lo, const bfx8 hi) {
    bfx16 r;
#pragma unroll
    for (int i = 0; i < 8; ++i) { r[i] = lo[i]; r[8 + i] = hi[i]; }
    return r;
}

__device__ __forceinline__ float gelu_exact(float x) {
    return 0.5f * x * (1.0f + erff(x * 0.70710678118654752f));
}

// ---------------- weight packing:  W[K,N] f32 -> Wt[N,K] bf16 ----------------
__global__ __launch_bounds__(256) void pack2d_kernel(const float* __restrict__ src,
                                                     __bf16* __restrict__ dst,
                                                     int K, int N) {
    int idx = blockIdx.x * 256 + threadIdx.x;
    if (idx >= K * N) return;
    int n = idx % N, k = idx / N;
    dst[(size_t)n * K + k] = (__bf16)src[idx];
}

// per-head weight (H, DIM, HEAD) -> Wt rows n = h*HEAD+d, cols k = c
__global__ __launch_bounds__(256) void pack_head_kernel(const float* __restrict__ src,
                                                        __bf16* __restrict__ dst) {
    int idx = blockIdx.x * 256 + threadIdx.x;
    if (idx >= HEADS * DIMC * HEADD) return;
    int d = idx % HEADD;
    int c = (idx / HEADD) % DIMC;
    int h = idx / (HEADD * DIMC);
    dst[(size_t)(h * HEADD + d) * DIMC + c] = (__bf16)src[idx];
}

// ---------------- LayerNorm -> bf16 ----------------
__global__ __launch_bounds__(256) void ln_bf16_kernel(const float* __restrict__ x,
                                                      const float* __restrict__ w,
                                                      const float* __restrict__ b,
                                                      __bf16* __restrict__ out,
                                                      float eps) {
    __shared__ float red[256];
    const int row = blockIdx.x, tid = threadIdx.x;
    const float* xr = x + (size_t)row * DIMC;
    float s = 0.f, s2 = 0.f;
    for (int c = tid; c < DIMC; c += 256) { float v = xr[c]; s += v; s2 += v * v; }
    red[tid] = s; __syncthreads();
    for (int off = 128; off > 0; off >>= 1) { if (tid < off) red[tid] += red[tid + off]; __syncthreads(); }
    float mu = red[0] * (1.0f / DIMC);
    __syncthreads();
    red[tid] = s2; __syncthreads();
    for (int off = 128; off > 0; off >>= 1) { if (tid < off) red[tid] += red[tid + off]; __syncthreads(); }
    float var = red[0] * (1.0f / DIMC) - mu * mu;
    float rs = rsqrtf(var + eps);
    for (int c = tid; c < DIMC; c += 256)
        out[(size_t)row * DIMC + c] = (__bf16)((xr[c] - mu) * rs * w[c] + b[c]);
}

// ---------------- generic WMMA GEMM:  C[M,N] = A[M,K](bf16) * Wt[N,K](bf16) ----------------
// wave tile 32(M) x 64(N): 2 A frags x 4 B frags -> 8 wmma per 32-wide k-step,
// double-buffered so next-step fragments are in flight during the wmma chain.
// EPI: 0 = f32 out; 1 = f32 out + res; 2 = bf16 out gelu(acc+bias);
//      3 = f32 out + bias + res;       4 = f32 out = res + scale[n]*(acc+bias)
template <int EPI>
__global__ __launch_bounds__(128) void gemm_wmma_kernel(const __bf16* __restrict__ A,
                                                        const __bf16* __restrict__ Wt,
                                                        const float* __restrict__ bias,
                                                        const float* __restrict__ res,
                                                        const float* __restrict__ scale,
                                                        float* __restrict__ outF,
                                                        __bf16* __restrict__ outB,
                                                        int M, int N, int K) {
    const int lane = threadIdx.x & 31;
    const int wave = threadIdx.x >> 5;
    const int m0 = blockIdx.y * 128 + wave * 32;   // 32 rows per wave
    const int n0 = blockIdx.x * 64;
    const int col = lane & 15;
    const int kbA = (lane < 16) ? 0 : 16;
    const int kbB = (lane < 16) ? 0 : 16;

    const __bf16* arow0 = A + (size_t)(m0 + col) * K + kbA;
    const __bf16* arow1 = arow0 + (size_t)16 * K;
    const __bf16* brow  = Wt + (size_t)(n0 + col) * K + kbB;

    v8f acc[8] = {};   // [h*4 + j]

    // prologue: fragments for k0 = 0
    bfx16 a_cur[2], b_cur[4];
    a_cur[0] = mkfrag(*(const bfx8*)(arow0), *(const bfx8*)(arow0 + 8));
    a_cur[1] = mkfrag(*(const bfx8*)(arow1), *(const bfx8*)(arow1 + 8));
#pragma unroll
    for (int j = 0; j < 4; ++j)
        b_cur[j] = *(const bfx16*)(brow + (size_t)j * 16 * K);

    for (int k0 = 0; k0 < K; k0 += 32) {
        const int k1 = k0 + 32;
        bfx16 a_nxt[2] = {a_cur[0], a_cur[1]};
        bfx16 b_nxt[4] = {b_cur[0], b_cur[1], b_cur[2], b_cur[3]};
        if (k1 < K) {
            __builtin_prefetch(arow0 + k1 + 256, 0, 1);
            __builtin_prefetch(brow + k1 + 256, 0, 1);
            a_nxt[0] = mkfrag(*(const bfx8*)(arow0 + k1), *(const bfx8*)(arow0 + k1 + 8));
            a_nxt[1] = mkfrag(*(const bfx8*)(arow1 + k1), *(const bfx8*)(arow1 + k1 + 8));
#pragma unroll
            for (int j = 0; j < 4; ++j)
                b_nxt[j] = *(const bfx16*)(brow + (size_t)j * 16 * K + k1);
        }
#pragma unroll
        for (int j = 0; j < 4; ++j) {
            acc[j]     = wmma_bf16(a_cur[0], b_cur[j], acc[j]);
            acc[4 + j] = wmma_bf16(a_cur[1], b_cur[j], acc[4 + j]);
        }
        a_cur[0] = a_nxt[0];
        a_cur[1] = a_nxt[1];
#pragma unroll
        for (int j = 0; j < 4; ++j) b_cur[j] = b_nxt[j];
    }

    const int halfrow = (lane >> 4) * 8;
#pragma unroll
    for (int hh = 0; hh < 2; ++hh) {
#pragma unroll
        for (int j = 0; j < 4; ++j) {
            const int n = n0 + j * 16 + col;
#pragma unroll
            for (int r = 0; r < 8; ++r) {
                const int m = m0 + hh * 16 + halfrow + r;
                const size_t o = (size_t)m * N + n;
                float v = acc[hh * 4 + j][r];
                if constexpr (EPI == 0) outF[o] = v;
                else if constexpr (EPI == 1) outF[o] = v + res[o];
                else if constexpr (EPI == 2) outB[o] = (__bf16)gelu_exact(v + bias[n]);
                else if constexpr (EPI == 3) outF[o] = v + bias[n] + res[o];
                else if constexpr (EPI == 4) outF[o] = res[o] + scale[n] * (v + bias[n]);
            }
        }
    }
}

// ---------------- xpos rotary: qkv f32 [M,1536] -> q,k bf16 [M,DIM] ----------------
__global__ __launch_bounds__(256) void rotary_kernel(const float* __restrict__ qkv,
                                                     __bf16* __restrict__ qbf,
                                                     __bf16* __restrict__ kbf) {
    int idx = blockIdx.x * 256 + threadIdx.x;      // over M * DIM/2 pairs
    if (idx >= MTOT * (DIMC / 2)) return;
    int p = idx % (DIMC / 2);
    int m = idx / (DIMC / 2);
    int t = m % TT;
    int c0 = 2 * p;
    int j = (c0 & (HEADD - 1)) >> 1;               // rotary index within head (0..63)

    float invf = __powf(10000.0f, -(float)j / (HEADD / 2));
    float ang = (float)t * invf;
    float cs = cosf(ang), sn = sinf(ang);
    float sv = (2.0f * j + 0.4f * HEADD) / (1.4f * HEADD);
    float sc = __powf(sv, (float)t / 512.0f);

    const size_t base = (size_t)m * (3 * DIMC);
    float q0 = qkv[base + c0],            q1 = qkv[base + c0 + 1];
    float k0 = qkv[base + DIMC + c0],     k1 = qkv[base + DIMC + c0 + 1];

    float cq = cs * sc, sq = sn * sc;
    float ck = cs / sc, sk = sn / sc;
    qbf[(size_t)m * DIMC + c0]     = (__bf16)(q0 * cq - q1 * sq);
    qbf[(size_t)m * DIMC + c0 + 1] = (__bf16)(q1 * cq + q0 * sq);
    kbf[(size_t)m * DIMC + c0]     = (__bf16)(k0 * ck - k1 * sk);
    kbf[(size_t)m * DIMC + c0 + 1] = (__bf16)(k1 * ck + k0 * sk);
}

// ---------------- v transpose: qkv[:,1024:1536] f32 -> vT[b,h,d,t] bf16 ----------------
__global__ __launch_bounds__(256) void vtrans_kernel(const float* __restrict__ qkv,
                                                     __bf16* __restrict__ vT) {
    int idx = blockIdx.x * 256 + threadIdx.x;      // over M * DIM
    if (idx >= MTOT * DIMC) return;
    int c = idx % DIMC;
    int m = idx / DIMC;
    int t = m % TT, b = m / TT;
    int h = c >> 7, d = c & 127;
    vT[(((size_t)b * HEADS + h) * HEADD + d) * TT + t] =
        (__bf16)qkv[(size_t)m * (3 * DIMC) + 2 * DIMC + c];
}

// ---------------- retention attention + group-norm ----------------
// one wave per (b, h, 16-row tile); causal decay mask g^(t-s)
__global__ __launch_bounds__(32) void attn_kernel(const __bf16* __restrict__ qbf,
                                                  const __bf16* __restrict__ kbf,
                                                  const __bf16* __restrict__ vT,
                                                  const float* __restrict__ gn_w,
                                                  const float* __restrict__ gn_b,
                                                  float* __restrict__ yout) {
    const int b = blockIdx.z, h = blockIdx.y;
    const int t0 = blockIdx.x * 16;
    const int lane = threadIdx.x;

    __shared__ __align__(16) __bf16 att_lds[16 * 32];
    __shared__ float y_lds[16 * HEADD];
    __shared__ float mu_s[16], rs_s[16];

    const float lng0 = logf(1.0f / 32.0f), lng1 = logf(1.0f / 512.0f);
    const float g = 1.0f - expf(lng0 + (float)h * (lng1 - lng0) / (HEADS - 1));
    const float log2g = log2f(g);

    const int col = lane & 15;
    const int halfrow = (lane >> 4) * 8;
    const int kbA = (lane < 16) ? 0 : 16;
    const int kbB = (lane < 16) ? 0 : 16;

    // preload q fragments (K = d, 4 chunks of 32)
    bfx16 qf[4];
    {
        const __bf16* qrow = qbf + ((size_t)(b * TT + t0 + col)) * DIMC + h * HEADD + kbA;
#pragma unroll
        for (int kt = 0; kt < 4; ++kt) {
            const bfx8 lo = *(const bfx8*)(qrow + kt * 32);
            const bfx8 hi = *(const bfx8*)(qrow + kt * 32 + 8);
            qf[kt] = mkfrag(lo, hi);
        }
    }

    v8f yacc[8] = {};

    for (int s0 = 0; s0 <= t0; s0 += 32) {
        const __bf16* k0p = kbf + ((size_t)(b * TT + s0 + col)) * DIMC + h * HEADD + kbB;
        const __bf16* k1p = k0p + (size_t)16 * DIMC;

        // batch-load all 8 k fragments so one wait covers the whole QK^T chain
        bfx16 bk0[4], bk1[4];
#pragma unroll
        for (int kt = 0; kt < 4; ++kt) {
            bk0[kt] = *(const bfx16*)(k0p + kt * 32);
            bk1[kt] = *(const bfx16*)(k1p + kt * 32);
        }
        v8f a0 = {}, a1 = {};
#pragma unroll
        for (int kt = 0; kt < 4; ++kt) {
            a0 = wmma_bf16(qf[kt], bk0[kt], a0);
            a1 = wmma_bf16(qf[kt], bk1[kt], a1);
        }

        // decay + causal mask, convert to bf16 A fragment via LDS
#pragma unroll
        for (int r = 0; r < 8; ++r) {
            const int t = t0 + halfrow + r;
            const int sa = s0 + col, sb = s0 + 16 + col;
            const float va = (sa <= t) ? a0[r] * exp2f(log2g * (float)(t - sa)) : 0.0f;
            const float vb = (sb <= t) ? a1[r] * exp2f(log2g * (float)(t - sb)) : 0.0f;
            att_lds[(halfrow + r) * 32 + col]      = (__bf16)va;
            att_lds[(halfrow + r) * 32 + 16 + col] = (__bf16)vb;
        }
        __syncthreads();

        // batch-load all 8 v fragments (overlaps the LDS round-trip)
        const __bf16* vp = vT + (((size_t)b * HEADS + h) * HEADD + col) * TT + s0 + kbB;
        bfx16 bv[8];
#pragma unroll
        for (int j = 0; j < 8; ++j)
            bv[j] = *(const bfx16*)(vp + (size_t)j * 16 * TT);

        bfx16 aatt;
        {
            const __bf16* ar = att_lds + col * 32 + kbA;
            const bfx8 lo = *(const bfx8*)ar;
            const bfx8 hi = *(const bfx8*)(ar + 8);
            aatt = mkfrag(lo, hi);
        }
        __syncthreads();

#pragma unroll
        for (int j = 0; j < 8; ++j)
            yacc[j] = wmma_bf16(aatt, bv[j], yacc[j]);
    }

    // group norm over HEAD=128 per row
#pragma unroll
    for (int j = 0; j < 8; ++j)
#pragma unroll
        for (int r = 0; r < 8; ++r)
            y_lds[(halfrow + r) * HEADD + j * 16 + col] = yacc[j][r];
    __syncthreads();
    if (lane < 16) {
        float s = 0.f, s2 = 0.f;
        for (int d = 0; d < HEADD; ++d) { float v = y_lds[lane * HEADD + d]; s += v; s2 += v * v; }
        float mu = s * (1.0f / HEADD);
        float var = s2 * (1.0f / HEADD) - mu * mu;
        mu_s[lane] = mu;
        rs_s[lane] = rsqrtf(var + 1e-5f);
    }
    __syncthreads();
#pragma unroll
    for (int j = 0; j < 8; ++j)
#pragma unroll
        for (int r = 0; r < 8; ++r) {
            const int t = t0 + halfrow + r;
            const int d = j * 16 + col;
            float v = (yacc[j][r] - mu_s[halfrow + r]) * rs_s[halfrow + r];
            v = v * gn_w[h * HEADD + d] + gn_b[h * HEADD + d];
            yout[((size_t)(b * TT + t)) * DIMC + h * HEADD + d] = v;
        }
}

// ---------------- z = silu(g) * y_norm  (bf16) ----------------
__global__ __launch_bounds__(256) void gate_kernel(const float* __restrict__ gbuf,
                                                   const float* __restrict__ ynorm,
                                                   __bf16* __restrict__ z) {
    int idx = blockIdx.x * 256 + threadIdx.x;
    if (idx >= MTOT * DIMC) return;
    float gg = gbuf[idx];
    z[idx] = (__bf16)(gg / (1.0f + expf(-gg)) * ynorm[idx]);
}

// ---------------- depthwise conv over T (+bias) ----------------
__global__ __launch_bounds__(256) void dwconv_kernel(const float* __restrict__ x,
                                                     const float* __restrict__ w,
                                                     const float* __restrict__ bias,
                                                     float* __restrict__ out, int ks) {
    int idx = blockIdx.x * 256 + threadIdx.x;
    if (idx >= MTOT * DIMC) return;
    int c = idx % DIMC;
    int m = idx / DIMC;
    int t = m % TT, b = m / TT;
    int pad = ks / 2;
    float acc = bias[c];
    for (int u = 0; u < ks; ++u) {
        int tt = t + u - pad;
        if (tt >= 0 && tt < TT)
            acc += x[((size_t)b * TT + tt) * DIMC + c] * w[c * ks + u];
    }
    out[idx] = acc;
}

// ---------------- host-side orchestration ----------------
extern "C" void kernel_launch(void* const* d_in, const int* in_sizes, int n_in,
                              void* d_out, int out_size, void* d_ws, size_t ws_size,
                              hipStream_t stream) {
    const float* x_in   = (const float*)d_in[0];
    const float* ln1_w  = (const float*)d_in[2];
    const float* ln1_b  = (const float*)d_in[3];
    const float* ln2_w  = (const float*)d_in[4];
    const float* ln2_b  = (const float*)d_in[5];
    const float* wq     = (const float*)d_in[6];
    const float* wk     = (const float*)d_in[7];
    const float* wv     = (const float*)d_in[8];
    const float* wg     = (const float*)d_in[9];
    const float* wo     = (const float*)d_in[10];
    const float* gn_w   = (const float*)d_in[11];
    const float* gn_b   = (const float*)d_in[12];
    const float* fw1    = (const float*)d_in[13];
    const float* fb1    = (const float*)d_in[14];
    const float* fw2    = (const float*)d_in[15];
    const float* fb2    = (const float*)d_in[16];
    const float* dw_w[3] = {(const float*)d_in[17], (const float*)d_in[18], (const float*)d_in[19]};
    const float* dw_b   = (const float*)d_in[20];
    const float* cn_nw  = (const float*)d_in[21];
    const float* cn_nb  = (const float*)d_in[22];
    const float* cw1    = (const float*)d_in[23];
    const float* cb1    = (const float*)d_in[24];
    const float* cw2    = (const float*)d_in[25];
    const float* cb2    = (const float*)d_in[26];
    const float* cn_g   = (const float*)d_in[27];

    // workspace carve-out (256B aligned)
    char* p = (char*)d_ws;
    auto alloc = [&](size_t bytes) -> void* {
        void* r = (void*)p;
        p += (bytes + 255) & ~(size_t)255;
        return r;
    };
    const size_t MD = (size_t)MTOT * DIMC;
    float*  xbuf   = (float*)alloc(MD * 4);
    __bf16* xn_bf  = (__bf16*)alloc(MD * 2);
    float*  qkvf   = (float*)alloc((size_t)MTOT * 3 * DIMC * 4);
    __bf16* q_bf   = (__bf16*)alloc(MD * 2);
    __bf16* k_bf   = (__bf16*)alloc(MD * 2);
    __bf16* vT_bf  = (__bf16*)alloc(MD * 2);
    float*  ynorm  = (float*)alloc(MD * 4);
    float*  gbuf   = (float*)alloc(MD * 4);
    __bf16* z_bf   = (__bf16*)alloc(MD * 2);
    float*  y1     = (float*)alloc(MD * 4);
    __bf16* h_bf   = (__bf16*)alloc(MD * 2);
    __bf16* mid_bf = (__bf16*)alloc((size_t)MTOT * FFNC * 2);
    float*  xt     = (float*)alloc(MD * 4);
    __bf16* xtn_bf = (__bf16*)alloc(MD * 2);
    __bf16* qkv_wt = (__bf16*)alloc((size_t)3 * DIMC * DIMC * 2);
    __bf16* wg_wt  = (__bf16*)alloc((size_t)DIMC * DIMC * 2);
    __bf16* wo_wt  = (__bf16*)alloc((size_t)DIMC * DIMC * 2);
    __bf16* fw1t   = (__bf16*)alloc((size_t)DIMC * FFNC * 2);
    __bf16* fw2t   = (__bf16*)alloc((size_t)DIMC * FFNC * 2);
    __bf16* cw1t   = (__bf16*)alloc((size_t)DIMC * FFNC * 2);
    __bf16* cw2t   = (__bf16*)alloc((size_t)DIMC * FFNC * 2);

    const int ks_arr[3] = {3, 7, 11};
    const int ew_grid  = (MTOT * DIMC + 255) / 256;
    const int rot_grid = (MTOT * (DIMC / 2) + 255) / 256;
    const int hw_grid  = (HEADS * DIMC * HEADD + 255) / 256;
    const int w2_grid  = (DIMC * DIMC + 255) / 256;
    const int wf_grid  = (DIMC * FFNC + 255) / 256;

    for (int i = 0; i < 3; ++i) {
        const float* xsrc = (i == 0) ? x_in : xbuf;

        // ---- pack this layer's weights into bf16 [N,K] ----
        pack_head_kernel<<<hw_grid, 256, 0, stream>>>(wq + (size_t)i * HEADS * DIMC * HEADD, qkv_wt);
        pack_head_kernel<<<hw_grid, 256, 0, stream>>>(wk + (size_t)i * HEADS * DIMC * HEADD, qkv_wt + (size_t)DIMC * DIMC);
        pack_head_kernel<<<hw_grid, 256, 0, stream>>>(wv + (size_t)i * HEADS * DIMC * HEADD, qkv_wt + (size_t)2 * DIMC * DIMC);
        pack2d_kernel<<<w2_grid, 256, 0, stream>>>(wg + (size_t)i * DIMC * DIMC, wg_wt, DIMC, DIMC);
        pack2d_kernel<<<w2_grid, 256, 0, stream>>>(wo + (size_t)i * DIMC * DIMC, wo_wt, DIMC, DIMC);
        pack2d_kernel<<<wf_grid, 256, 0, stream>>>(fw1 + (size_t)i * DIMC * FFNC, fw1t, DIMC, FFNC);
        pack2d_kernel<<<wf_grid, 256, 0, stream>>>(fw2 + (size_t)i * FFNC * DIMC, fw2t, FFNC, DIMC);
        pack2d_kernel<<<wf_grid, 256, 0, stream>>>(cw1 + (size_t)i * DIMC * FFNC, cw1t, DIMC, FFNC);
        pack2d_kernel<<<wf_grid, 256, 0, stream>>>(cw2 + (size_t)i * FFNC * DIMC, cw2t, FFNC, DIMC);

        // ---- ln1 -> xn (bf16) ----
        ln_bf16_kernel<<<MTOT, 256, 0, stream>>>(xsrc, ln1_w + i * DIMC, ln1_b + i * DIMC, xn_bf, 1e-5f);

        // ---- fused qkv projection: [8192,1536] ----
        gemm_wmma_kernel<0><<<dim3((3 * DIMC) / 64, MTOT / 128), 128, 0, stream>>>(
            xn_bf, qkv_wt, nullptr, nullptr, nullptr, qkvf, nullptr, MTOT, 3 * DIMC, DIMC);

        rotary_kernel<<<rot_grid, 256, 0, stream>>>(qkvf, q_bf, k_bf);
        vtrans_kernel<<<ew_grid, 256, 0, stream>>>(qkvf, vT_bf);

        // ---- retention attention + group norm ----
        attn_kernel<<<dim3(TT / 16, HEADS, BB), 32, 0, stream>>>(
            q_bf, k_bf, vT_bf, gn_w + i * DIMC, gn_b + i * DIMC, ynorm);

        // ---- gate: g = xn @ wg ; z = silu(g) * y ----
        gemm_wmma_kernel<0><<<dim3(DIMC / 64, MTOT / 128), 128, 0, stream>>>(
            xn_bf, wg_wt, nullptr, nullptr, nullptr, gbuf, nullptr, MTOT, DIMC, DIMC);
        gate_kernel<<<ew_grid, 256, 0, stream>>>(gbuf, ynorm, z_bf);

        // ---- y1 = z @ wo + x ----
        gemm_wmma_kernel<1><<<dim3(DIMC / 64, MTOT / 128), 128, 0, stream>>>(
            z_bf, wo_wt, nullptr, xsrc, nullptr, y1, nullptr, MTOT, DIMC, DIMC);

        // ---- FFN ----
        ln_bf16_kernel<<<MTOT, 256, 0, stream>>>(y1, ln2_w + i * DIMC, ln2_b + i * DIMC, h_bf, 1e-5f);
        gemm_wmma_kernel<2><<<dim3(FFNC / 64, MTOT / 128), 128, 0, stream>>>(
            h_bf, fw1t, fb1 + i * FFNC, nullptr, nullptr, nullptr, mid_bf, MTOT, FFNC, DIMC);
        gemm_wmma_kernel<3><<<dim3(DIMC / 64, MTOT / 128), 128, 0, stream>>>(
            mid_bf, fw2t, fb2 + i * DIMC, y1, nullptr, xbuf, nullptr, MTOT, DIMC, FFNC);

        // ---- conv branch ----
        dwconv_kernel<<<ew_grid, 256, 0, stream>>>(xbuf, dw_w[i], dw_b + i * DIMC, xt, ks_arr[i]);
        ln_bf16_kernel<<<MTOT, 256, 0, stream>>>(xt, cn_nw + i * DIMC, cn_nb + i * DIMC, xtn_bf, 1e-6f);
        gemm_wmma_kernel<2><<<dim3(FFNC / 64, MTOT / 128), 128, 0, stream>>>(
            xtn_bf, cw1t, cb1 + i * FFNC, nullptr, nullptr, nullptr, mid_bf, MTOT, FFNC, DIMC);

        float* xdst = (i == 2) ? (float*)d_out : xbuf;
        gemm_wmma_kernel<4><<<dim3(DIMC / 64, MTOT / 128), 128, 0, stream>>>(
            mid_bf, cw2t, cb2 + i * DIMC, xbuf, cn_g + i * DIMC, xdst, nullptr, MTOT, DIMC, FFNC);
    }
}
